// MultiHeadAttention_21775484191277
// MI455X (gfx1250) — matrix operand order repaired and checked by reading, not compiled, not run on hardware
//
#include <hip/hip_runtime.h>

#define Bsz     2
#define Sseq    2048
#define NH      16
#define HD      64
#define HIDDEN  1024
#define MROWS   4096           // B*S
#define SCALE_F 0.125f         // 1/sqrt(64)

typedef __bf16 bf16_t;
typedef __attribute__((ext_vector_type(16))) __bf16 v16bf;
typedef __attribute__((ext_vector_type(8)))  __bf16 v8bf;
typedef __attribute__((ext_vector_type(8)))  float  v8f;
typedef __attribute__((ext_vector_type(4)))  float  v4f;
typedef __attribute__((ext_vector_type(4)))  int    v4i;

// ---- optional CDNA5 async global->LDS path (ASYNCcnt) --------------------
#if defined(__has_builtin)
#  if __has_builtin(__builtin_amdgcn_global_load_async_to_lds_b128) && \
      __has_builtin(__builtin_amdgcn_s_wait_asynccnt)
#    define USE_ASYNC_LDS 1
#  endif
#endif
#ifndef USE_ASYNC_LDS
#  define USE_ASYNC_LDS 0
#endif

__device__ __forceinline__ void lds_copy16(const bf16_t* g, bf16_t* l) {
#if USE_ASYNC_LDS
    // signature: (global v4i*, shared v4i*, imm offset, imm cpol)
    __builtin_amdgcn_global_load_async_to_lds_b128(
        (__attribute__((address_space(1))) v4i*)(g),
        (__attribute__((address_space(3))) v4i*)(l), 0, 0);
#else
    *(v8bf*)l = *(const v8bf*)g;
#endif
}

__device__ __forceinline__ void async_fence() {
#if USE_ASYNC_LDS
    __builtin_amdgcn_s_wait_asynccnt(0);
#endif
}

union FragBF { v16bf v; v8bf h[2]; };

__device__ __forceinline__ v8f zero8() {
    v8f z = {0.f, 0.f, 0.f, 0.f, 0.f, 0.f, 0.f, 0.f};
    return z;
}

__device__ __forceinline__ v8f wmma_bf16(v16bf a, v16bf b, v8f c) {
    // (neg_a, A, neg_b, B, c_mod, C, reuse_a, reuse_b)
    return __builtin_amdgcn_wmma_f32_16x16x32_bf16(false, a, false, b,
                                                   (short)0, c, false, false);
}

__device__ __forceinline__ v16bf frag2(const bf16_t* p0, const bf16_t* p1) {
    FragBF f;
    f.h[0] = *(const v8bf*)p0;
    f.h[1] = *(const v8bf*)p1;
    return f.v;
}

// ---------------------------------------------------------------------------
// Tiled GEMM: C[M=4096, N=1024] = A[4096,1024] * W[1024,1024]^T + bias
// BM=BN=128, BK=32, double-buffered LDS. 256 threads = 8 waves;
// each wave computes 2x4 16x16 WMMA tiles per K-step.
// ---------------------------------------------------------------------------
template<bool A_IS_BF16, bool OUT_IS_F32>
__global__ __launch_bounds__(256)
void gemm128_wmma(const void* __restrict__ Aptr,
                  const float* __restrict__ W,
                  const float* __restrict__ bias,
                  void* __restrict__ Cptr)
{
    __shared__ __align__(16) bf16_t sA[2][128 * 32];
    __shared__ __align__(16) bf16_t sB[2][128 * 32];

    const int tid  = threadIdx.x;
    const int lane = tid & 31;
    const int wave = tid >> 5;
    const int wm   = wave >> 1;          // 0..3 -> m-tile pair
    const int wn   = wave & 1;           // 0..1 -> n-tile quad
    const int l15  = lane & 15;
    const int hs   = lane >> 4;
    const int m0   = blockIdx.y * 128;
    const int n0   = blockIdx.x * 128;

    v8f acc[2][4];
    #pragma unroll
    for (int i = 0; i < 2; ++i)
        #pragma unroll
        for (int j = 0; j < 4; ++j)
            acc[i][j] = zero8();

    auto stage = [&](int kt, int buf) {
        const int k0 = kt * 32;
        if constexpr (A_IS_BF16) {
            const bf16_t* A = (const bf16_t*)Aptr;
            #pragma unroll
            for (int i = 0; i < 2; ++i) {
                int e = (i * 256 + tid) * 8;
                int r = e >> 5, c = e & 31;
                lds_copy16(A + (m0 + r) * HIDDEN + k0 + c,
                           &sA[buf][r * 32 + c]);
            }
        } else {
            const float* A = (const float*)Aptr;
            #pragma unroll
            for (int i = 0; i < 4; ++i) {
                int e = (i * 256 + tid) * 4;
                int r = e >> 5, c = e & 31;
                v4f x = *(const v4f*)(A + (m0 + r) * HIDDEN + k0 + c);
                sA[buf][r * 32 + c + 0] = (bf16_t)x.x;
                sA[buf][r * 32 + c + 1] = (bf16_t)x.y;
                sA[buf][r * 32 + c + 2] = (bf16_t)x.z;
                sA[buf][r * 32 + c + 3] = (bf16_t)x.w;
            }
        }
        #pragma unroll
        for (int i = 0; i < 4; ++i) {
            int e = (i * 256 + tid) * 4;
            int r = e >> 5, c = e & 31;
            v4f x = *(const v4f*)(W + (n0 + r) * HIDDEN + k0 + c);
            sB[buf][r * 32 + c + 0] = (bf16_t)x.x;
            sB[buf][r * 32 + c + 1] = (bf16_t)x.y;
            sB[buf][r * 32 + c + 2] = (bf16_t)x.z;
            sB[buf][r * 32 + c + 3] = (bf16_t)x.w;
        }
    };

    const int NT = HIDDEN / 32;
    stage(0, 0);
    async_fence();
    __syncthreads();

    for (int kt = 0; kt < NT; ++kt) {
        const int buf = kt & 1;
        if (kt + 1 < NT)                     // prefetch next tile into other buf
            stage(kt + 1, buf ^ 1);

        v16bf af[2];
        #pragma unroll
        for (int i = 0; i < 2; ++i) {
            const bf16_t* base = &sA[buf][(wm * 2 + i) * 16 * 32];
            int o0 = l15 * 32 + hs * 8;              // A: two 8-half runs
            af[i] = frag2(base + o0, base + o0 + 16);
        }
        v16bf bfr[4];
        #pragma unroll
        for (int j = 0; j < 4; ++j) {
            const bf16_t* base = &sB[buf][(wn * 4 + j) * 16 * 32];
            int o0 = l15 * 32 + hs * 8;              // B: two 8-half runs
            bfr[j] = frag2(base + o0, base + o0 + 16);
        }
        #pragma unroll
        for (int i = 0; i < 2; ++i)
            #pragma unroll
            for (int j = 0; j < 4; ++j)
                acc[i][j] = wmma_bf16(af[i], bfr[j], acc[i][j]);

        async_fence();
        __syncthreads();
    }

    // ---- epilogue: bias + store ----
    #pragma unroll
    for (int j = 0; j < 4; ++j) {
        const int col = n0 + (wn * 4 + j) * 16 + l15;
        const float bj = bias[col];
        #pragma unroll
        for (int i = 0; i < 2; ++i) {
            #pragma unroll
            for (int r = 0; r < 8; ++r) {
                const int row = m0 + (wm * 2 + i) * 16 + r + 8 * hs;
                const float v = acc[i][j][r] + bj;
                if constexpr (OUT_IS_F32)
                    ((float*)Cptr)[(size_t)row * HIDDEN + col] = v;
                else
                    ((bf16_t*)Cptr)[(size_t)row * HIDDEN + col] = (bf16_t)v;
            }
        }
    }
}

// ---------------------------------------------------------------------------
// Causal flash attention. Q/K/V bf16 [B*S, HID] (col = h*64 + d).
// Block = (128 q-rows, one head, one batch); 8 waves, 16 q-rows per wave.
// KV blocks of 64, double-buffered; K staged row-major (async path when
// available), V staged transposed [d][kv] in LDS.
// ---------------------------------------------------------------------------
__global__ __launch_bounds__(256)
void flash_attn_wmma(const bf16_t* __restrict__ Q,
                     const bf16_t* __restrict__ Kp,
                     const bf16_t* __restrict__ Vp,
                     bf16_t* __restrict__ Oout)
{
    __shared__ __align__(16) bf16_t sK [2][64 * 64];   // [kv][d]
    __shared__ __align__(16) bf16_t sVT[2][64 * 64];   // [d][kv]
    __shared__ __align__(16) bf16_t sP [8][16 * 64];   // per-wave P tile

    const int tid  = threadIdx.x;
    const int lane = tid & 31;
    const int wave = tid >> 5;
    const int l15  = lane & 15;
    const int hs   = lane >> 4;

    const int qb = blockIdx.x;                 // q-block (128 rows)
    const int h  = blockIdx.y;
    const int b  = blockIdx.z;
    const int rbase   = b * Sseq;
    const int q0      = qb * 128 + wave * 16;  // this wave's first q row
    const int colbase = h * HD;

    // Q tile (16 x 64) kept in registers as two A-fragments over d
    v16bf aq[2];
    {
        const bf16_t* qrow = Q + (size_t)(rbase + q0 + l15) * HIDDEN + colbase;
        #pragma unroll
        for (int c = 0; c < 2; ++c) {
            int o0 = c * 32 + hs * 8;
            aq[c] = frag2(qrow + o0, qrow + o0 + 16);
        }
    }

    float rowmax[8], rowsum[8];
    v8f accO[4];
    #pragma unroll
    for (int r = 0; r < 8; ++r) { rowmax[r] = -3.0e38f; rowsum[r] = 0.f; }
    #pragma unroll
    for (int j = 0; j < 4; ++j) accO[j] = zero8();

    bf16_t* sPw = &sP[wave][0];
    const int nkv = qb * 2 + 2;                // 64-wide kv blocks needed

    auto stage_kv = [&](int kvb, int buf) {
        const int kv0 = kvb * 64;
        #pragma unroll
        for (int i = 0; i < 2; ++i) {
            int e = (i * 256 + tid) * 8;
            int r = e >> 6, c = e & 63;
            const size_t g = (size_t)(rbase + kv0 + r) * HIDDEN + colbase + c;
            lds_copy16(Kp + g, &sK[buf][r * 64 + c]);
            v8bf vv = *(const v8bf*)(Vp + g);
            #pragma unroll
            for (int j = 0; j < 8; ++j)
                sVT[buf][(c + j) * 64 + r] = vv[j];
        }
    };

    stage_kv(0, 0);
    async_fence();
    __syncthreads();

    for (int kvb = 0; kvb < nkv; ++kvb) {
        const int kv0 = kvb * 64;
        const int buf = kvb & 1;
        if (kvb + 1 < nkv)                 // prefetch next kv block
            stage_kv(kvb + 1, buf ^ 1);

        if (kv0 <= q0 + 15) {              // wave-uniform causal skip
            // ---- scores: S = Q * K^T (8 WMMAs) ----
            v8f sc[4];
            #pragma unroll
            for (int nt = 0; nt < 4; ++nt) {
                sc[nt] = zero8();
                const bf16_t* kb = &sK[buf][nt * 16 * 64];
                #pragma unroll
                for (int c = 0; c < 2; ++c) {
                    int o0 = l15 * 64 + c * 32 + hs * 8;
                    v16bf bk = frag2(kb + o0, kb + o0 + 16);
                    sc[nt] = wmma_bf16(aq[c], bk, sc[nt]);
                }
            }
            const bool need_mask = (kv0 + 63) > q0;
            // ---- online softmax (per C-frag row r + 8*hs) ----
            #pragma unroll
            for (int r = 0; r < 8; ++r) {
                const int qrow = q0 + r + 8 * hs;
                float mx = -3.0e38f;
                #pragma unroll
                for (int nt = 0; nt < 4; ++nt) {
                    float v = sc[nt][r] * SCALE_F;
                    if (need_mask) {
                        const int kvi = kv0 + nt * 16 + l15;
                        v = (kvi > qrow) ? -3.0e38f : v;
                    }
                    sc[nt][r] = v;
                    mx = fmaxf(mx, v);
                }
                #pragma unroll
                for (int off = 8; off >= 1; off >>= 1)
                    mx = fmaxf(mx, __shfl_xor(mx, off, 32));
                const float mnew  = fmaxf(rowmax[r], mx);
                const float scale = __expf(rowmax[r] - mnew);
                rowmax[r] = mnew;
                float ps = 0.f;
                #pragma unroll
                for (int nt = 0; nt < 4; ++nt) {
                    const float p = __expf(sc[nt][r] - mnew);
                    ps += p;
                    sPw[(r + 8 * hs) * 64 + nt * 16 + l15] = (bf16_t)p;
                }
                #pragma unroll
                for (int off = 8; off >= 1; off >>= 1)
                    ps += __shfl_xor(ps, off, 32);
                rowsum[r] = rowsum[r] * scale + ps;
                #pragma unroll
                for (int j = 0; j < 4; ++j)
                    accO[j][r] *= scale;
            }
            // ---- O += P * V (re-read P as A-fragments; 8 WMMAs) ----
            v16bf ap[2];
            #pragma unroll
            for (int c = 0; c < 2; ++c) {
                int o0 = l15 * 64 + c * 32 + hs * 8;
                ap[c] = frag2(sPw + o0, sPw + o0 + 16);
            }
            #pragma unroll
            for (int j = 0; j < 4; ++j) {
                const bf16_t* vb = &sVT[buf][j * 16 * 64];
                #pragma unroll
                for (int c = 0; c < 2; ++c) {
                    int o0 = l15 * 64 + c * 32 + hs * 8;
                    v16bf bv = frag2(vb + o0, vb + o0 + 16);
                    accO[j] = wmma_bf16(ap[c], bv, accO[j]);
                }
            }
        }

        async_fence();
        __syncthreads();
    }

    // ---- normalize + store bf16 attention output ----
    #pragma unroll
    for (int r = 0; r < 8; ++r) {
        const float inv = 1.0f / rowsum[r];
        const int row = rbase + q0 + r + 8 * hs;
        #pragma unroll
        for (int j = 0; j < 4; ++j)
            Oout[(size_t)row * HIDDEN + colbase + j * 16 + l15] =
                (bf16_t)(accO[j][r] * inv);
    }
}

// ---------------------------------------------------------------------------
extern "C" void kernel_launch(void* const* d_in, const int* in_sizes, int n_in,
                              void* d_out, int out_size, void* d_ws, size_t ws_size,
                              hipStream_t stream)
{
    (void)in_sizes; (void)n_in; (void)out_size; (void)ws_size;

    const float* x  = (const float*)d_in[0];
    const float* Wq = (const float*)d_in[1];
    const float* Wk = (const float*)d_in[2];
    const float* Wv = (const float*)d_in[3];
    const float* Wo = (const float*)d_in[4];
    const float* bq = (const float*)d_in[5];
    const float* bk = (const float*)d_in[6];
    const float* bv = (const float*)d_in[7];
    const float* bo = (const float*)d_in[8];

    const size_t seg = (size_t)MROWS * HIDDEN;   // bf16 elements per tensor
    bf16_t* qws = (bf16_t*)d_ws;
    bf16_t* kws = qws + seg;
    bf16_t* vws = kws + seg;
    bf16_t* aws = vws + seg;

    dim3 gGemm(HIDDEN / 128, MROWS / 128);
    gemm128_wmma<false, false><<<gGemm, 256, 0, stream>>>(x, Wq, bq, qws);
    gemm128_wmma<false, false><<<gGemm, 256, 0, stream>>>(x, Wk, bk, kws);
    gemm128_wmma<false, false><<<gGemm, 256, 0, stream>>>(x, Wv, bv, vws);

    dim3 gAttn(Sseq / 128, NH, Bsz);
    flash_attn_wmma<<<gAttn, 256, 0, stream>>>(qws, kws, vws, aws);

    gemm128_wmma<true, true><<<gGemm, 256, 0, stream>>>(aws, Wo, bo, d_out);
}
